// MultiHeadAttention_75213467287740
// MI455X (gfx1250) — matrix operand order repaired and checked by reading, not compiled, not run on hardware
//
#include <hip/hip_runtime.h>
#include <math.h>

// MI455X / gfx1250: wave32, f16 WMMA 16x16x32 with fp32 accumulation.
typedef __attribute__((ext_vector_type(16))) _Float16 v16h;
typedef __attribute__((ext_vector_type(8)))  float    v8f;
typedef __attribute__((ext_vector_type(4)))  unsigned int v4u;

union AF { v16h v; v4u q[2]; };

#define BB 4
#define NN 2048
#define EE 768
#define HH 12
#define DD 64
#define K3E 2304
#define ATT_SCALE 0.125f   // 1/sqrt(64)

__device__ __forceinline__ v8f wmma16(const AF& a, const AF& b, v8f c) {
  return __builtin_amdgcn_wmma_f32_16x16x32_f16(
      false, a.v, false, b.v, (short)0, c, false, false);
}

// ---- CDNA5 async global->LDS path (ASYNCcnt-tracked, no VGPR round trip) ----
__device__ __forceinline__ void async_ld_b128(unsigned lds_off, const _Float16* g) {
  // VDST = per-lane LDS byte offset, VADDR = per-lane 64-bit global address (GV mode)
  asm volatile("global_load_async_to_lds_b128 %0, %1, off"
               :: "v"(lds_off), "v"(g) : "memory");
}
template <int N>
__device__ __forceinline__ void wait_async() {
#if __has_builtin(__builtin_amdgcn_s_wait_asynccnt)
  __builtin_amdgcn_s_wait_asynccnt(N);
#else
  asm volatile("s_wait_asynccnt %0" :: "n"(N) : "memory");
#endif
}

// ---------------------------------------------------------------------------
// Prep 1: x (fp32) -> xh (f16), straight copy/convert.
// ---------------------------------------------------------------------------
__global__ __launch_bounds__(256) void k_cvt(const float* __restrict__ in,
                                             _Float16* __restrict__ out, int n) {
  const int i = (blockIdx.x * 256 + threadIdx.x) * 4;
  if (i < n) {
    float4 v = *(const float4*)&in[i];
    out[i]     = (_Float16)v.x;
    out[i + 1] = (_Float16)v.y;
    out[i + 2] = (_Float16)v.z;
    out[i + 3] = (_Float16)v.w;
  }
}

// ---------------------------------------------------------------------------
// Prep 2: W [K][N] fp32 -> Wt [N][K] f16 (LDS tile transpose, coalesced both sides).
// Pre-transposed f16 weights make every B-fragment a contiguous-k global b128 run.
// ---------------------------------------------------------------------------
__global__ __launch_bounds__(256) void k_tw(const float* __restrict__ in,
                                            _Float16* __restrict__ out,
                                            int K, int N) {
  __shared__ float tile[32][33];
  const int nb = N >> 5;
  const int bx = blockIdx.x % nb;   // n-tile
  const int by = blockIdx.x / nb;   // k-tile
  const int tx = threadIdx.x & 31, ty = threadIdx.x >> 5;  // 32x8
#pragma unroll
  for (int p = 0; p < 32; p += 8)
    tile[ty + p][tx] = in[(size_t)(by * 32 + ty + p) * N + bx * 32 + tx];
  __syncthreads();
#pragma unroll
  for (int p = 0; p < 32; p += 8)
    out[(size_t)(bx * 32 + ty + p) * K + by * 32 + tx] = (_Float16)tile[tx][ty + p];
}

// ---------------------------------------------------------------------------
// Kernel 1: qkv = xh @ Wqkv + bqkv, 32x64 tile per wave, operands straight
// from global as b128 fragments (A reused x4, B reused x2). Scatter into
// Q[b,h,n,d], K[b,h,n,d], Vt[b,h,d,n] as f16.
// ---------------------------------------------------------------------------
__global__ __launch_bounds__(256) void k_qkv(const _Float16* __restrict__ xh,
                                             const _Float16* __restrict__ Wt,  // [2304][768]
                                             const float* __restrict__ bqkv,
                                             _Float16* __restrict__ Qb,
                                             _Float16* __restrict__ Kb,
                                             _Float16* __restrict__ Vt) {
  const int lane = threadIdx.x & 31;
  const int gw = blockIdx.x * 8 + (threadIdx.x >> 5);   // 9216 waves
  const int mt = gw / 36, nt = gw % 36;                 // 256 x 36 tiles
  const int m0 = mt * 32, n0 = nt * 64;
  const int ln = lane & 15, hi = lane >> 4;
  const int aoff = hi ? 8 : 0, koff = hi ? 16 : 0;

  v8f acc[2][4] = {};
  for (int k0 = 0; k0 < EE; k0 += 32) {
    AF a[2];
#pragma unroll
    for (int i = 0; i < 2; ++i) {
      const _Float16* ar = xh + (size_t)(m0 + i * 16 + ln) * EE + k0;
      a[i].q[0] = *(const v4u*)(ar + koff);
      a[i].q[1] = *(const v4u*)(ar + koff + 8);
    }
#pragma unroll
    for (int j = 0; j < 4; ++j) {
      AF b;
      const _Float16* br = Wt + (size_t)(n0 + j * 16 + ln) * EE + k0;
      b.q[0] = *(const v4u*)(br + koff);
      b.q[1] = *(const v4u*)(br + koff + 8);
      acc[0][j] = wmma16(a[0], b, acc[0][j]);
      acc[1][j] = wmma16(a[1], b, acc[1][j]);
    }
    if (k0 + 32 < EE) __builtin_prefetch(xh + (size_t)(m0 + ln) * EE + k0 + 32, 0, 1);
  }

#pragma unroll
  for (int j = 0; j < 4; ++j) {
    const int c = n0 + j * 16 + ln;   // column in [0,2304): which=c%3, d=(c/3)%64, h=c/192
    const float bias = bqkv[c];
    const int which = c % 3;
    const int rem = c / 3;
    const int d = rem & 63;
    const int h = rem >> 6;
#pragma unroll
    for (int i = 0; i < 2; ++i)
#pragma unroll
      for (int r = 0; r < 8; ++r) {
        const int mg = m0 + i * 16 + r + hi * 8;
        const int bb = mg >> 11;
        const int nn = mg & (NN - 1);
        const _Float16 val = (_Float16)(acc[i][j][r] + bias);
        if (which == 2) {
          Vt[(((size_t)(bb * HH + h)) * DD + d) * NN + nn] = val;
        } else {
          _Float16* dst = (which == 0) ? Qb : Kb;
          dst[(((size_t)(bb * HH + h)) * NN + nn) * DD + d] = val;
        }
      }
  }
}

// ---------------------------------------------------------------------------
// Kernel 2: flash attention. All 8 waves of a block share one (b,h); K/V
// chunks (64 keys) are staged cooperatively into LDS with CDNA5 async
// global->LDS b128 copies, double-buffered on ASYNCcnt (each wave keeps its
// next-chunk 4 ops in flight: s_wait_asynccnt 4). 16 WMMAs per 64 keys.
// ---------------------------------------------------------------------------
__global__ __launch_bounds__(256) void k_attn(const _Float16* __restrict__ Qb,
                                              const _Float16* __restrict__ Kb,
                                              const _Float16* __restrict__ Vt,
                                              _Float16* __restrict__ AO) {
  __shared__ __align__(16) _Float16 Klds[2][64 * 64];   // [t][d] 8KB x2
  __shared__ __align__(16) _Float16 Vlds[2][64 * 64];   // [d][t] 8KB x2
  __shared__ __align__(16) _Float16 Plds[8][16 * 64];   // per-wave P tile
  const int lane = threadIdx.x & 31;
  const int w = threadIdx.x >> 5;
  const int ln = lane & 15, hi = lane >> 4;
  const int bh = blockIdx.x >> 4;                 // 16 blocks per (b,h)
  const int h = bh % HH, b = bh / HH;
  const int qt = (blockIdx.x & 15) * 8 + w;
  const int q0 = qt * 16;
  const _Float16* Qg = Qb + (size_t)(b * HH + h) * NN * DD;
  const _Float16* Kg = Kb + (size_t)(b * HH + h) * NN * DD;
  const _Float16* Vg = Vt + (size_t)(b * HH + h) * DD * NN;
  const int aoff = hi ? 8 : 0, koff = hi ? 16 : 0;

  const unsigned kbase = (unsigned)(unsigned long long)&Klds[0][0];
  const unsigned vbase = (unsigned)(unsigned long long)&Vlds[0][0];

  auto issue = [&](int t0, int buf) {
    const unsigned kb = kbase + (unsigned)buf * 8192u;
    const unsigned vb = vbase + (unsigned)buf * 8192u;
#pragma unroll
    for (int c = (int)threadIdx.x; c < 512; c += 256) {
      // K chunk is one contiguous 8KB block (rows t0..t0+63 of [t][d])
      async_ld_b128(kb + c * 16, Kg + (size_t)t0 * DD + c * 8);
      // V chunk: 64 row-slices of 128B from [d][t] (stride NN)
      const int r = c >> 3, cc = c & 7;
      async_ld_b128(vb + c * 16, Vg + (size_t)r * NN + t0 + cc * 8);
    }
  };

  AF aq[2];   // Q fragments, reused across all key chunks
#pragma unroll
  for (int kd2 = 0; kd2 < 2; ++kd2) {
    const _Float16* qr = Qg + (size_t)(q0 + ln) * DD + kd2 * 32;
    aq[kd2].q[0] = *(const v4u*)(qr + koff);
    aq[kd2].q[1] = *(const v4u*)(qr + koff + 8);
  }

  float mrow[8], lrow[8];
#pragma unroll
  for (int r = 0; r < 8; ++r) { mrow[r] = -__builtin_inff(); lrow[r] = 0.f; }
  v8f o[4] = {};

  issue(0, 0);
  for (int ch = 0; ch < NN / 64; ++ch) {
    const int buf = ch & 1;
    if (ch + 1 < NN / 64) { issue((ch + 1) * 64, buf ^ 1); wait_async<4>(); }
    else                  { wait_async<0>(); }
    __syncthreads();                      // chunk ch data visible block-wide
    const _Float16* Kl = &Klds[buf][0];
    const _Float16* Vl = &Vlds[buf][0];

    v8f s[4] = {};
#pragma unroll
    for (int kd2 = 0; kd2 < 2; ++kd2)
#pragma unroll
      for (int jt = 0; jt < 4; ++jt) {    // S = Q K^T, 4 key tiles of 16
        AF bk;
        const _Float16* kr = Kl + (jt * 16 + ln) * DD + kd2 * 32 + koff;
        bk.q[0] = *(const v4u*)(kr);
        bk.q[1] = *(const v4u*)(kr + 8);
        s[jt] = wmma16(aq[kd2], bk, s[jt]);
      }

    // Online softmax over 64 columns (rows 0-7 in lanes 0-15, 8-15 in 16-31)
    float p[4][8];
#pragma unroll
    for (int r = 0; r < 8; ++r) {
      float v0 = s[0][r] * ATT_SCALE, v1 = s[1][r] * ATT_SCALE;
      float v2 = s[2][r] * ATT_SCALE, v3 = s[3][r] * ATT_SCALE;
      float mx = fmaxf(fmaxf(v0, v1), fmaxf(v2, v3));
#pragma unroll
      for (int msk = 1; msk < 16; msk <<= 1) mx = fmaxf(mx, __shfl_xor(mx, msk, 32));
      const float mnew = fmaxf(mrow[r], mx);
      v0 = __expf(v0 - mnew); v1 = __expf(v1 - mnew);
      v2 = __expf(v2 - mnew); v3 = __expf(v3 - mnew);
      float rs = (v0 + v1) + (v2 + v3);
#pragma unroll
      for (int msk = 1; msk < 16; msk <<= 1) rs += __shfl_xor(rs, msk, 32);
      const float corr = __expf(mrow[r] - mnew);   // first chunk: exp(-inf)=0
      lrow[r] = lrow[r] * corr + rs;
      mrow[r] = mnew;
#pragma unroll
      for (int j = 0; j < 4; ++j) o[j][r] *= corr;
      p[0][r] = v0; p[1][r] = v1; p[2][r] = v2; p[3][r] = v3;
    }

#pragma unroll
    for (int r = 0; r < 8; ++r) {         // C-layout -> per-wave LDS P tile
      const int row = r + hi * 8;
#pragma unroll
      for (int jt = 0; jt < 4; ++jt)
        Plds[w][row * 64 + jt * 16 + ln] = (_Float16)p[jt][r];
    }
    // same-wave LDS ops are in-order; DScnt wait inserted by compiler
#pragma unroll
    for (int s2 = 0; s2 < 2; ++s2) {      // O += P @ V (K-dim 64 = 2x32)
      AF ap;
      const _Float16* pr = &Plds[w][ln * 64 + s2 * 32];
      ap.q[0] = *(const v4u*)(pr + koff);
      ap.q[1] = *(const v4u*)(pr + koff + 8);
#pragma unroll
      for (int j = 0; j < 4; ++j) {
        AF bv;
        const _Float16* vr = Vl + (j * 16 + ln) * DD + s2 * 32 + koff;
        bv.q[0] = *(const v4u*)(vr);
        bv.q[1] = *(const v4u*)(vr + 8);
        o[j] = wmma16(ap, bv, o[j]);
      }
    }
    __syncthreads();   // all LDS reads done before next issue overwrites buf
  }

#pragma unroll
  for (int j = 0; j < 4; ++j)
#pragma unroll
    for (int r = 0; r < 8; ++r) {
      const int m = q0 + r + hi * 8;
      AO[((size_t)(b * NN + m)) * EE + h * DD + j * 16 + ln] =
          (_Float16)(o[j][r] / lrow[r]);
    }
}

// ---------------------------------------------------------------------------
// Kernel 3: out = AO @ Wproj + bproj (fp32 out), 32x64 tile per wave,
// pure global-fragment GEMM (Wproj pre-transposed f16).
// ---------------------------------------------------------------------------
__global__ __launch_bounds__(256) void k_proj(const _Float16* __restrict__ AO,
                                              const _Float16* __restrict__ Wt,  // [768][768]
                                              const float* __restrict__ bp,
                                              float* __restrict__ out) {
  const int lane = threadIdx.x & 31;
  const int gw = blockIdx.x * 8 + (threadIdx.x >> 5);   // 3072 waves
  const int mt = gw / 12, nt = gw % 12;                 // 256 x 12 tiles
  const int m0 = mt * 32, n0 = nt * 64;
  const int ln = lane & 15, hi = lane >> 4;
  const int aoff = hi ? 8 : 0, koff = hi ? 16 : 0;

  v8f acc[2][4] = {};
  for (int k0 = 0; k0 < EE; k0 += 32) {
    AF a[2];
#pragma unroll
    for (int i = 0; i < 2; ++i) {
      const _Float16* ar = AO + (size_t)(m0 + i * 16 + ln) * EE + k0;
      a[i].q[0] = *(const v4u*)(ar + koff);
      a[i].q[1] = *(const v4u*)(ar + koff + 8);
    }
#pragma unroll
    for (int j = 0; j < 4; ++j) {
      AF b;
      const _Float16* br = Wt + (size_t)(n0 + j * 16 + ln) * EE + k0;
      b.q[0] = *(const v4u*)(br + koff);
      b.q[1] = *(const v4u*)(br + koff + 8);
      acc[0][j] = wmma16(a[0], b, acc[0][j]);
      acc[1][j] = wmma16(a[1], b, acc[1][j]);
    }
    if (k0 + 32 < EE) __builtin_prefetch(AO + (size_t)(m0 + ln) * EE + k0 + 32, 0, 1);
  }
#pragma unroll
  for (int j = 0; j < 4; ++j) {
    const float bias = bp[n0 + j * 16 + ln];
#pragma unroll
    for (int i = 0; i < 2; ++i)
#pragma unroll
      for (int r = 0; r < 8; ++r)
        out[(size_t)(m0 + i * 16 + r + hi * 8) * EE + n0 + j * 16 + ln] =
            acc[i][j][r] + bias;
  }
}

// ---------------------------------------------------------------------------
extern "C" void kernel_launch(void* const* d_in, const int* in_sizes, int n_in,
                              void* d_out, int out_size, void* d_ws, size_t ws_size,
                              hipStream_t stream) {
  const float* x     = (const float*)d_in[0];
  const float* Wqkv  = (const float*)d_in[1];
  const float* bqkv  = (const float*)d_in[2];
  const float* Wproj = (const float*)d_in[3];
  const float* bproj = (const float*)d_in[4];
  float* out = (float*)d_out;

  // Workspace (all f16): xh, Wqkv^T, Wproj^T, Q, K, Vt, AO  (~67 MB)
  const size_t nX = (size_t)BB * NN * EE;         // 6,291,456
  const size_t nQ = (size_t)BB * HH * NN * DD;    // 6,291,456
  _Float16* xh  = (_Float16*)d_ws;
  _Float16* Wqt = xh + nX;                        // [2304][768]
  _Float16* Wpt = Wqt + (size_t)K3E * EE;         // [768][768]
  _Float16* Qb  = Wpt + (size_t)EE * EE;
  _Float16* Kb  = Qb + nQ;
  _Float16* Vt  = Kb + nQ;
  _Float16* AO  = Vt + nQ;

  k_cvt<<<dim3((unsigned)(nX / 4 / 256)), dim3(256), 0, stream>>>(x, xh, (int)nX);
  k_tw <<<dim3((EE / 32) * (K3E / 32)), dim3(256), 0, stream>>>(Wqkv, Wqt, EE, K3E);
  k_tw <<<dim3((EE / 32) * (EE / 32)),  dim3(256), 0, stream>>>(Wproj, Wpt, EE, EE);

  k_qkv <<<dim3(256 * 36 / 8), dim3(256), 0, stream>>>(xh, Wqt, bqkv, Qb, Kb, Vt);
  k_attn<<<dim3(BB * HH * 16), dim3(256), 0, stream>>>(Qb, Kb, Vt, AO);
  k_proj<<<dim3(256 * 12 / 8), dim3(256), 0, stream>>>(AO, Wpt, bproj, out);

  (void)in_sizes; (void)n_in; (void)out_size; (void)ws_size;
}
